// DGP_RF_Embeddings_1056561955054
// MI455X (gfx1250) — matrix operand order repaired and checked by reading, not compiled, not run on hardware
//
#include <hip/hip_runtime.h>

typedef __attribute__((ext_vector_type(16))) _Float16 v16h;
typedef __attribute__((ext_vector_type(8)))  float    v8f;

struct __align__(16) U4 { unsigned x, y, z, w; };

union FragH {
  v16h h;
  unsigned u[8];
};

__device__ __forceinline__ unsigned packf2(float a, float b) {
  union { _Float16 h[2]; unsigned u; } t;
  t.h[0] = (_Float16)a;
  t.h[1] = (_Float16)b;
  return t.u;
}

__device__ __forceinline__ void loadB32(FragH& f, const _Float16* p) {
  // 16 consecutive halves (32 bytes) -> two b128 loads
  U4 a = *(const U4*)(p);
  U4 b = *(const U4*)(p + 8);
  f.u[0] = a.x; f.u[1] = a.y; f.u[2] = a.z; f.u[3] = a.w;
  f.u[4] = b.x; f.u[5] = b.y; f.u[6] = b.z; f.u[7] = b.w;
}

__device__ __forceinline__ void loadA_lds(FragH& f, const _Float16* p) {
  // halves [0..7] and [16..23] relative to p
  U4 a = *(const U4*)(p);
  U4 b = *(const U4*)(p + 16);
  f.u[0] = a.x; f.u[1] = a.y; f.u[2] = a.z; f.u[3] = a.w;
  f.u[4] = b.x; f.u[5] = b.y; f.u[6] = b.z; f.u[7] = b.w;
}

// ---------------- zero accumulators ----------------
__global__ void vb_zero(float* __restrict__ p, int n) {
  int i = blockIdx.x * blockDim.x + threadIdx.x;
  if (i < n) p[i] = 0.0f;
}

// ---------------- weight prep: f32 -> f16 transposed ----------------
__global__ void vb_prep(const float* __restrict__ W1mu, const float* __restrict__ W1lv,
                        const float* __restrict__ b1lv,
                        const float* __restrict__ W2mu, const float* __restrict__ W2lv,
                        const float* __restrict__ b2lv,
                        _Float16* __restrict__ W1Tmu, _Float16* __restrict__ W1Tvar,
                        _Float16* __restrict__ W2Tmu, _Float16* __restrict__ W2Tmsv,
                        _Float16* __restrict__ W2Tvar,
                        float* __restrict__ b1v, float* __restrict__ b2v) {
  int gid = blockIdx.x * blockDim.x + threadIdx.x;  // 0..65535
  {
    int r = gid >> 7, k = gid & 127;                // W1: [128][512] -> W1T[r][k]
    float mu  = W1mu[k * 512 + r];
    float var = __expf(W1lv[k * 512 + r]);
    W1Tmu[r * 128 + k]  = (_Float16)mu;
    W1Tvar[r * 128 + k] = (_Float16)var;
  }
  {
    int d = gid >> 9, k = gid & 511;                // W2: [512][128] -> W2T[d][k]
    float mu  = W2mu[k * 128 + d];
    float var = __expf(W2lv[k * 128 + d]);
    W2Tmu[d * 512 + k]  = (_Float16)mu;
    W2Tmsv[d * 512 + k] = (_Float16)(mu * mu + var);
    W2Tvar[d * 512 + k] = (_Float16)var;
  }
  if (gid < 512) b1v[gid] = __expf(b1lv[gid]);
  if (gid < 128) b2v[gid] = __expf(b2lv[gid]);
}

// ---------------- fused VB-MLP + segment atomics ----------------
__global__ __launch_bounds__(64) void vb_main(
    const float* __restrict__ X, const long long* __restrict__ Xidx,
    const float* __restrict__ b1mu, const float* __restrict__ b2mu,
    const float* __restrict__ b1v, const float* __restrict__ b2v,
    const _Float16* __restrict__ W1Tmu, const _Float16* __restrict__ W1Tvar,
    const _Float16* __restrict__ W2Tmu, const _Float16* __restrict__ W2Tmsv,
    const _Float16* __restrict__ W2Tvar,
    float* __restrict__ meanS, float* __restrict__ wS) {
  __shared__ __align__(16) _Float16 shE[2][16 * 512];
  __shared__ __align__(16) _Float16 shV[2][16 * 512];

  const int lane = threadIdx.x & 31;
  const int wave = threadIdx.x >> 5;
  const int hh   = lane >> 4;   // lane half: 0/1
  const int nn   = lane & 15;   // column / row-within-tile index
  const int rowBase = (blockIdx.x * 2 + wave) * 16;

  _Float16* He = shE[wave];
  _Float16* Hv = shV[wave];

  // ---- build layer-1 A fragments for X and X^2 (16x128 tile, f16) ----
  FragH Am[4], Av[4];
#pragma unroll
  for (int kb = 0; kb < 4; ++kb) {
    const float* xp = X + (size_t)(rowBase + nn) * 128 + kb * 32 + hh * 8;
    const float4* fp = (const float4*)xp;
    float4 f0 = fp[0], f1 = fp[1];      // K offsets 0..7
    float4 g0 = fp[4], g1 = fp[5];      // K offsets 16..23
    Am[kb].u[0] = packf2(f0.x, f0.y); Am[kb].u[1] = packf2(f0.z, f0.w);
    Am[kb].u[2] = packf2(f1.x, f1.y); Am[kb].u[3] = packf2(f1.z, f1.w);
    Am[kb].u[4] = packf2(g0.x, g0.y); Am[kb].u[5] = packf2(g0.z, g0.w);
    Am[kb].u[6] = packf2(g1.x, g1.y); Am[kb].u[7] = packf2(g1.z, g1.w);
    Av[kb].u[0] = packf2(f0.x*f0.x, f0.y*f0.y); Av[kb].u[1] = packf2(f0.z*f0.z, f0.w*f0.w);
    Av[kb].u[2] = packf2(f1.x*f1.x, f1.y*f1.y); Av[kb].u[3] = packf2(f1.z*f1.z, f1.w*f1.w);
    Av[kb].u[4] = packf2(g0.x*g0.x, g0.y*g0.y); Av[kb].u[5] = packf2(g0.z*g0.z, g0.w*g0.w);
    Av[kb].u[6] = packf2(g1.x*g1.x, g1.y*g1.y); Av[kb].u[7] = packf2(g1.z*g1.z, g1.w*g1.w);
  }

  // ---- layer 1: D=128 -> R=512, ReLU moment matching, stash e/ve in LDS ----
  for (int ct = 0; ct < 32; ++ct) {
    v8f accM = {};
    v8f accV = {};
    const int r = ct * 16 + nn;
#pragma unroll
    for (int kb = 0; kb < 4; ++kb) {
      FragH Bm, Bv;
      loadA_lds(Bm, W1Tmu  + r * 128 + kb * 32 + hh * 8);
      loadA_lds(Bv, W1Tvar + r * 128 + kb * 32 + hh * 8);
      accM = __builtin_amdgcn_wmma_f32_16x16x32_f16(false, Am[kb].h, false, Bm.h,
                                                    (short)0, accM, false, false);
      accV = __builtin_amdgcn_wmma_f32_16x16x32_f16(false, Av[kb].h, false, Bv.h,
                                                    (short)0, accV, false, false);
    }
    const float bm = b1mu[r];
    const float bv = b1v[r];
#pragma unroll
    for (int r8 = 0; r8 < 8; ++r8) {
      float m  = accM[r8] + bm;
      float vv = accV[r8] + bv;
      // fast reciprocal-sqrt path: s = v*rs, a = m*rs  (no v_sqrt/v_div sequences)
      float rs = __builtin_amdgcn_rsqf(fmaxf(vv, 1e-12f));
      float s  = vv * rs;
      float a  = m * rs;
      // branch-free Gaussian pdf + cdf (Zelen-Severo, |err| < 7.5e-8):
      // phi(a) shared between pdf and the cdf tail.
      float q   = __expf(-0.5f * a * a);
      float pdf = 0.39894228040143270f * q;
      float ax  = fabsf(a);
      float kk  = __builtin_amdgcn_rcpf(fmaf(0.2316419f, ax, 1.0f));
      float poly = kk * fmaf(kk,
                     fmaf(kk, fmaf(kk, fmaf(kk, 1.330274429f, -1.821255978f),
                                       1.781477937f),
                              -0.356563782f),
                     0.319381530f);
      float tail = pdf * poly;
      float cdf  = (a >= 0.0f) ? (1.0f - tail) : tail;
      float sp  = s * pdf;
      float e   = fmaf(m, cdf, sp);
      float e2  = fmaf(fmaf(m, m, vv), cdf, m * sp);
      float ve  = fmaxf(e2 - e * e, 0.0f);
      int M = r8 + 8 * hh;
      He[M * 512 + r] = (_Float16)e;
      Hv[M * 512 + r] = (_Float16)ve;
    }
  }
  __syncthreads();

  // segment ids for my 8 rows (C/D layout rows: M = r8 + 8*half)
  int seg[8];
#pragma unroll
  for (int r8 = 0; r8 < 8; ++r8)
    seg[r8] = (int)Xidx[rowBase + r8 + 8 * hh];

  // ---- layer 2: R=512 -> D=128, then precision-weighted segment atomics ----
  for (int ct2 = 0; ct2 < 8; ++ct2) {
    v8f accM = {};
    v8f accV = {};
    const int d = ct2 * 16 + nn;
    for (int kb = 0; kb < 16; ++kb) {
      FragH Ae, Ave, Ae2, Bm, Bs, Bv;
      const int abase = nn * 512 + kb * 32 + hh * 8;
      loadA_lds(Ae,  He + abase);
      loadA_lds(Ave, Hv + abase);
      Ae2.h = Ae.h * Ae.h;  // packed f16 square: e^2 fragment
      loadA_lds(Bm, W2Tmu  + d * 512 + kb * 32 + hh * 8);
      loadA_lds(Bs, W2Tmsv + d * 512 + kb * 32 + hh * 8);
      loadA_lds(Bv, W2Tvar + d * 512 + kb * 32 + hh * 8);
      accM = __builtin_amdgcn_wmma_f32_16x16x32_f16(false, Ae.h,  false, Bm.h,
                                                    (short)0, accM, false, false);
      accV = __builtin_amdgcn_wmma_f32_16x16x32_f16(false, Ave.h, false, Bs.h,
                                                    (short)0, accV, false, false);
      accV = __builtin_amdgcn_wmma_f32_16x16x32_f16(false, Ae2.h, false, Bv.h,
                                                    (short)0, accV, false, false);
    }
    const float bm2 = b2mu[d];
    const float bv2 = b2v[d];
#pragma unroll
    for (int r8 = 0; r8 < 8; ++r8) {
      float m  = accM[r8] + bm2;
      float vv = accV[r8] + bv2;
      float p  = __builtin_amdgcn_rcpf(fmaxf(vv, 1e-8f));  // native v_rcp_f32
      float* wp = wS    + (size_t)seg[r8] * 128 + d;
      float* mp = meanS + (size_t)seg[r8] * 128 + d;
      unsafeAtomicAdd(wp, p);          // native global_atomic_add_f32
      unsafeAtomicAdd(mp, p * m);
    }
  }
}

// ---------------- finalize: means/vars per segment ----------------
__global__ void vb_final(const float* __restrict__ meanS, const float* __restrict__ wS,
                         float* __restrict__ out, int SD) {
  int i = blockIdx.x * blockDim.x + threadIdx.x;
  if (i < SD) {
    float inv = 1.0f / (wS[i] + 1e-8f);
    out[i]      = meanS[i] * inv;  // embedd_means
    out[SD + i] = inv;             // embedd_vars
  }
}

extern "C" void kernel_launch(void* const* d_in, const int* in_sizes, int n_in,
                              void* d_out, int out_size, void* d_ws, size_t ws_size,
                              hipStream_t stream) {
  const float*     X    = (const float*)d_in[0];
  const long long* Xidx = (const long long*)d_in[1];   // int64 indices
  const float* W1mu = (const float*)d_in[3];
  const float* W1lv = (const float*)d_in[4];
  const float* b1mu = (const float*)d_in[5];
  const float* b1lv = (const float*)d_in[6];
  const float* W2mu = (const float*)d_in[7];
  const float* W2lv = (const float*)d_in[8];
  const float* b2mu = (const float*)d_in[9];
  const float* b2lv = (const float*)d_in[10];

  const int N  = in_sizes[1];        // 131072
  const int SD = out_size / 2;       // S * 128

  // workspace layout
  float* meanS = (float*)d_ws;                       // SD f32
  float* wS    = meanS + SD;                         // SD f32
  _Float16* W1Tmu  = (_Float16*)(wS + SD);           // 512*128
  _Float16* W1Tvar = W1Tmu  + 512 * 128;
  _Float16* W2Tmu  = W1Tvar + 512 * 128;             // 128*512
  _Float16* W2Tmsv = W2Tmu  + 512 * 128;
  _Float16* W2Tvar = W2Tmsv + 512 * 128;
  float* b1v = (float*)(W2Tvar + 512 * 128);         // 512
  float* b2v = b1v + 512;                            // 128

  vb_zero<<<(2 * SD + 255) / 256, 256, 0, stream>>>(meanS, 2 * SD);
  vb_prep<<<256, 256, 0, stream>>>(W1mu, W1lv, b1lv, W2mu, W2lv, b2lv,
                                   W1Tmu, W1Tvar, W2Tmu, W2Tmsv, W2Tvar, b1v, b2v);
  vb_main<<<N / 32, 64, 0, stream>>>(X, Xidx, b1mu, b2mu, b1v, b2v,
                                     W1Tmu, W1Tvar, W2Tmu, W2Tmsv, W2Tvar,
                                     meanS, wS);
  vb_final<<<(SD + 255) / 256, 256, 0, stream>>>(meanS, wS, (float*)d_out, SD);
}
